// MaskLSTM_27865747816713
// MI455X (gfx1250) — matrix operand order repaired and checked by reading, not compiled, not run on hardware
//
#include <hip/hip_runtime.h>
#include <hip/hip_bf16.h>

typedef __bf16 bf16_t;
typedef __attribute__((ext_vector_type(16))) __bf16 v16bf;
typedef __attribute__((ext_vector_type(8)))  __bf16 v8bf;
typedef __attribute__((ext_vector_type(4)))  __bf16 v4bf;
typedef __attribute__((ext_vector_type(8)))  float  v8f;
typedef __attribute__((ext_vector_type(4)))  float  v4f;

#define WMMA_BF16(a, b, c) \
  __builtin_amdgcn_wmma_f32_16x16x32_bf16(false, (a), false, (b), (short)0, (c), false, false)

// Problem constants: B=64, L=1024, D=256, H=256, 4H=1024
__device__ inline float sigmoid_fast(float x) { return 1.f / (1.f + __expf(-x)); }
__device__ inline float tanh_fast(float x) {
  float ax = fabsf(x);
  float e  = __expf(-2.f * ax);
  float r  = (1.f - e) / (1.f + e);
  return x < 0.f ? -r : r;
}

// ---------------------------------------------------------------------------
// Kernel 1: convert weights fp32 -> bf16 (row-major [g][k] kept so WMMA
// B-fragments are contiguous 32B loads), and fuse b_ih + b_hh.
// ---------------------------------------------------------------------------
__global__ void cvt_kernel(const float* __restrict__ Wih, const float* __restrict__ Whh,
                           const float* __restrict__ bih, const float* __restrict__ bhh,
                           bf16_t* __restrict__ WihB, bf16_t* __restrict__ WhhB,
                           float* __restrict__ bsum) {
  const int i = blockIdx.x * blockDim.x + threadIdx.x;
  const int n = 1024 * 256;
  for (int idx = i; idx < n; idx += gridDim.x * blockDim.x) {
    WihB[idx] = (bf16_t)Wih[idx];
    WhhB[idx] = (bf16_t)Whh[idx];
  }
  if (i < 1024) bsum[i] = bih[i] + bhh[i];
}

// ---------------------------------------------------------------------------
// Kernel 2: xproj GEMM.  P[l*64+b][g] = sum_d x[b][l][d]*W_ih[g][d] + bias[g]
// M = 65536 rows (time-major), N = 1024, K = 256.
// Wave computes a 16x64 tile: A-fragment reused across 4 N-tiles.
// ---------------------------------------------------------------------------
__global__ __launch_bounds__(256)
void xproj_kernel(const float* __restrict__ x, const bf16_t* __restrict__ WihB,
                  const float* __restrict__ bsum, bf16_t* __restrict__ P) {
  const int lane   = threadIdx.x & 31;
  const int waveId = blockIdx.x * 8 + (threadIdx.x >> 5);
  const int mTile  = waveId >> 4;          // 0..4095
  const int n0     = (waveId & 15) * 64;   // 0..960
  const int m0     = mTile * 16;

  // per-lane A row:  m = l*64 + b  ->  x[b][l][:]
  const int r = lane & 15;
  const int m = m0 + r;
  const int l = m >> 6, b = m & 63;
  const float* arow = x + ((size_t)b * 1024 + l) * 256;
  const int kA = (lane & 16) ? 16 : 0;     // A-fragment K half (contiguous, matches B)

  const int nlane = lane & 15;
  const int kB = (lane & 16) ? 16 : 0;     // B-fragment K half
  const bf16_t* brow0 = WihB + (size_t)(n0 + nlane) * 256 + kB;

  v8f acc0 = {0,0,0,0,0,0,0,0}, acc1 = acc0, acc2 = acc0, acc3 = acc0;

  for (int k0 = 0; k0 < 256; k0 += 32) {
    const float* pa = arow + k0 + kA;
    if (k0 + 32 < 256) __builtin_prefetch(pa + 32, 0, 1);   // global_prefetch_b8
    v4bf a0 = __builtin_convertvector(*(const v4f*)(pa),      v4bf);
    v4bf a1 = __builtin_convertvector(*(const v4f*)(pa + 4),  v4bf);
    v4bf a2 = __builtin_convertvector(*(const v4f*)(pa + 8),  v4bf);
    v4bf a3 = __builtin_convertvector(*(const v4f*)(pa + 12), v4bf);
    v8bf lo = __builtin_shufflevector(a0, a1, 0,1,2,3,4,5,6,7);
    v8bf hi = __builtin_shufflevector(a2, a3, 0,1,2,3,4,5,6,7);
    v16bf af = __builtin_shufflevector(lo, hi, 0,1,2,3,4,5,6,7,8,9,10,11,12,13,14,15);

    v16bf bf0 = *(const v16bf*)(brow0 + (size_t)0 * 16 * 256 + k0);
    v16bf bf1 = *(const v16bf*)(brow0 + (size_t)1 * 16 * 256 + k0);
    v16bf bf2 = *(const v16bf*)(brow0 + (size_t)2 * 16 * 256 + k0);
    v16bf bf3 = *(const v16bf*)(brow0 + (size_t)3 * 16 * 256 + k0);
    acc0 = WMMA_BF16(af, bf0, acc0);
    acc1 = WMMA_BF16(af, bf1, acc1);
    acc2 = WMMA_BF16(af, bf2, acc2);
    acc3 = WMMA_BF16(af, bf3, acc3);
  }

  // epilogue: C/D layout -> lane = col + 16*(row>>3), vgpr v = row&7
  const int mrow = m0 + ((lane & 16) ? 8 : 0);
  v8f accs[4] = {acc0, acc1, acc2, acc3};
  #pragma unroll
  for (int nt = 0; nt < 4; ++nt) {
    const int g = n0 + nt * 16 + nlane;
    const float bias = bsum[g];
    #pragma unroll
    for (int v = 0; v < 8; ++v)
      P[(size_t)(mrow + v) * 1024 + g] = (bf16_t)(accs[nt][v] + bias);
  }
}

// ---------------------------------------------------------------------------
// Kernel 3: recurrent scan.  4 workgroups, each owns a 16-row batch slice
// entirely in LDS (no cross-WG sync; batch rows are independent recurrences).
// Per step: gates[16][1024] = h_bf16[16x256] @ W_hh^T via 64 WMMA tiles
// (16 waves x 4 N-tiles x 8 K-steps), then fused gate nonlinearity + mask
// blend.  g = 0..1023 forward, 1024..2047 backward (carry continues).
// ---------------------------------------------------------------------------
__global__ __launch_bounds__(512)
void scan_kernel(const bf16_t* __restrict__ P, const bf16_t* __restrict__ WhhB,
                 const float* __restrict__ mask, float* __restrict__ out,
                 float* __restrict__ hn, float* __restrict__ cn) {
  __shared__ __align__(32) bf16_t sHb[16 * 256];   // 8KB  h (bf16, WMMA A)
  __shared__ float sH[16 * 256];                    // 16KB h (fp32 state)
  __shared__ float sC[16 * 256];                    // 16KB c (fp32 state)
  __shared__ float sG[16 * 1024];                   // 64KB gates (fp32)

  const int tid  = threadIdx.x;
  const int b0   = blockIdx.x * 16;                 // batch slice base
  const int lane = tid & 31;
  const int wave = tid >> 5;                        // 0..15

  for (int i = tid; i < 16 * 256; i += 512) { sH[i] = 0.f; sC[i] = 0.f; sHb[i] = (bf16_t)0.f; }
  __syncthreads();

  const int nlane = lane & 15;
  const int kB = (lane & 16) ? 16 : 0;
  const int kA = (lane & 16) ? 16 : 0;
  const int rA = lane & 15;
  const int mrow = (lane & 16) ? 8 : 0;

  #pragma unroll 1
  for (int g = 0; g < 2048; ++g) {
    const int dir = g >> 10;
    const int t   = dir ? (2047 - g) : g;

    // ---- GEMM: sG = sHb @ W_hh^T  (M=16, N=1024, K=256) ----
    #pragma unroll
    for (int ntl = 0; ntl < 4; ++ntl) {
      const int n0 = (wave * 4 + ntl) * 16;
      const bf16_t* brow = WhhB + (size_t)(n0 + nlane) * 256 + kB;
      v8f acc = {0,0,0,0,0,0,0,0};
      #pragma unroll
      for (int k0 = 0; k0 < 256; k0 += 32) {
        const bf16_t* pa = &sHb[rA * 256 + k0 + kA];
        v8bf alo = *(const v8bf*)pa;                 // ds_load
        v8bf ahi = *(const v8bf*)(pa + 8);
        v16bf af = __builtin_shufflevector(alo, ahi, 0,1,2,3,4,5,6,7,8,9,10,11,12,13,14,15);
        v16bf bf = *(const v16bf*)(brow + k0);       // L0/L2-resident W_hh
        acc = WMMA_BF16(af, bf, acc);
      }
      #pragma unroll
      for (int v = 0; v < 8; ++v)
        sG[(mrow + v) * 1024 + n0 + nlane] = acc[v];
    }
    __syncthreads();

    // ---- fused gates + mask blend: 16x256 = 4096 elems, 8 per thread ----
    #pragma unroll
    for (int e = 0; e < 8; ++e) {
      const int idx = tid + e * 512;
      const int bl  = idx >> 8;        // local batch row
      const int j   = idx & 255;       // hidden unit
      const size_t prow = ((size_t)t * 64 + (b0 + bl)) * 1024;
      const float gi = sG[bl * 1024 + j      ] + (float)P[prow + j      ];
      const float gf = sG[bl * 1024 + 256 + j] + (float)P[prow + 256 + j];
      const float gg = sG[bl * 1024 + 512 + j] + (float)P[prow + 512 + j];
      const float go = sG[bl * 1024 + 768 + j] + (float)P[prow + 768 + j];
      const float mt = mask[(size_t)(b0 + bl) * 1024 + t];
      const float cp = sC[bl * 256 + j];
      const float hp = sH[bl * 256 + j];
      const float c1 = sigmoid_fast(gf) * cp + sigmoid_fast(gi) * tanh_fast(gg);
      const float h1 = sigmoid_fast(go) * tanh_fast(c1);
      const float hnew = mt * h1 + (1.f - mt) * hp;
      const float cnew = mt * c1 + (1.f - mt * cp);   // reference's parenthesization
      sH[bl * 256 + j]  = hnew;
      sC[bl * 256 + j]  = cnew;
      sHb[bl * 256 + j] = (bf16_t)hnew;
      out[(size_t)(b0 + bl) * (1024 * 512) + (size_t)t * 512 + dir * 256 + j] = hnew;
      if (g == 1023) { hn[(b0 + bl) * 512 + j] = hnew;       cn[(b0 + bl) * 512 + j] = cnew; }
      if (g == 2047) { hn[(b0 + bl) * 512 + 256 + j] = hnew; cn[(b0 + bl) * 512 + 256 + j] = cnew; }
    }
    __syncthreads();
  }
}

// ---------------------------------------------------------------------------
extern "C" void kernel_launch(void* const* d_in, const int* in_sizes, int n_in,
                              void* d_out, int out_size, void* d_ws, size_t ws_size,
                              hipStream_t stream) {
  const float* x    = (const float*)d_in[0];   // [64,1024,256]
  const float* mask = (const float*)d_in[1];   // [64,1024]
  const float* Wih  = (const float*)d_in[2];   // [1024,256]
  const float* Whh  = (const float*)d_in[3];   // [1024,256]
  const float* bih  = (const float*)d_in[4];   // [1024]
  const float* bhh  = (const float*)d_in[5];   // [1024]
  (void)in_sizes; (void)n_in; (void)out_size; (void)ws_size;  // bid==1 hardcoded

  char* ws = (char*)d_ws;
  size_t off = 0;
  bf16_t* P    = (bf16_t*)(ws + off); off += (size_t)65536 * 1024 * sizeof(bf16_t); // 134MB
  bf16_t* WihB = (bf16_t*)(ws + off); off += (size_t)1024 * 256 * sizeof(bf16_t);
  bf16_t* WhhB = (bf16_t*)(ws + off); off += (size_t)1024 * 256 * sizeof(bf16_t);
  float*  bsum = (float*) (ws + off); off += 1024 * sizeof(float);

  float* out = (float*)d_out;
  float* hn  = out + (size_t)64 * 1024 * 512;
  float* cn  = hn + 64 * 512;

  cvt_kernel  <<<dim3(512),  dim3(256), 0, stream>>>(Wih, Whh, bih, bhh, WihB, WhhB, bsum);
  xproj_kernel<<<dim3(8192), dim3(256), 0, stream>>>(x, WihB, bsum, P);
  scan_kernel <<<dim3(4),    dim3(512), 0, stream>>>(P, WhhB, mask, out, hn, cn);
}
